// DGCNN_Grouper_1967095021879
// MI455X (gfx1250) — compile-verified, matrix operand order checked by reading, not run on hardware
//
#include <hip/hip_runtime.h>

// ---------------- problem constants ----------------
constexpr int   B_    = 8;
constexpr int   N_    = 4096;
constexpr int   KNB   = 16;     // k nearest neighbors
constexpr int   CMIDH = 8;      // Cmid/2
constexpr int   COUT  = 128;
constexpr int   NP    = 1024;   // NPOINT
constexpr int   NG    = 4;      // groups
constexpr float EPS_  = 1e-5f;
constexpr float SLOPE = 0.2f;

// d_out layout (floats): [xyz B*3*NP][feat B*COUT*NP][idx B*NP]
constexpr size_t OUT_FEAT = (size_t)B_ * 3 * NP;                 // 24576
constexpr size_t OUT_IDX  = OUT_FEAT + (size_t)B_ * COUT * NP;   // 1073152

typedef __attribute__((ext_vector_type(2)))  float    v2f;
typedef __attribute__((ext_vector_type(8)))  float    v8f;
typedef __attribute__((ext_vector_type(16))) _Float16 v16h;

// ---------------------------------------------------------------------------
// 1) lift: fm[b][c][n] = w_in[c,:]·f[b,:,n] + b_in[c]   (c<8)
// ---------------------------------------------------------------------------
__global__ void k_lift(const float* __restrict__ f, const float* __restrict__ w_in,
                       const float* __restrict__ b_in, float* __restrict__ fm) {
  int gid = blockIdx.x * blockDim.x + threadIdx.x;
  if (gid >= B_ * N_) return;
  int b = gid >> 12, n = gid & (N_ - 1);
  float f0 = f[((size_t)(b * 3 + 0)) * N_ + n];
  float f1 = f[((size_t)(b * 3 + 1)) * N_ + n];
  float f2 = f[((size_t)(b * 3 + 2)) * N_ + n];
#pragma unroll
  for (int c = 0; c < CMIDH; c++) {
    float v = w_in[c * 3 + 0] * f0 + w_in[c * 3 + 1] * f1 + w_in[c * 3 + 2] * f2 + b_in[c];
    fm[((size_t)(b * CMIDH + c)) * N_ + n] = v;
  }
}

// ---------------------------------------------------------------------------
// 2) KNN via V_WMMA_F32_16X16X4_F32.  One wave per 16 queries.
//    key(q,p) = |p|^2 - 2 q·p  (query norm drops out of per-row ranking)
//    C layout: lane l holds col (l&15), rows r+8*(l>>4)  -> LDS transpose,
//    2 lanes per row scan 8 cols each keeping sorted top-16; merge at end.
//    Single-wave block: DS ops are in-order per wave, so a compiler-only
//    wave_barrier() suffices instead of s_barrier (which is a NOP here anyway).
// ---------------------------------------------------------------------------
__global__ void k_knn(const float* __restrict__ x, int* __restrict__ knn_out) {
  const int b    = blockIdx.y;
  const int q0   = blockIdx.x * 16;
  const int lane = threadIdx.x;          // 0..31 (one wave)
  const int ln   = lane & 15;
  const int hi   = lane >> 4;
  const float* xb = x + (size_t)b * 3 * N_;

  // A fragment (queries, 16x4): lanes 0-15 hold K=0,1 (x,y); lanes 16-31 K=2,3 (z,0)
  const int cA0 = hi ? 2 : 0;
  const int q   = q0 + ln;
  float aq0 = xb[cA0 * N_ + q];
  float aq1 = xb[N_ + q];                // always-valid load, then select
  v2f afrag;
  afrag.x = aq0;
  afrag.y = hi ? 0.0f : aq1;

  __shared__ float keytile[16][17];
  __shared__ float mgk[16][32];
  __shared__ int   mgi[16][32];

  float bestk[KNB];
  int   besti[KNB];
#pragma unroll
  for (int i = 0; i < KNB; i++) { bestk[i] = 3.0e38f; besti[i] = 0; }

  for (int m0 = 0; m0 < N_; m0 += 16) {
    const int p = m0 + ln;
    float b0  = xb[cA0 * N_ + p];
    float b1y = xb[N_ + p];              // unconditional load (valid), select below
    float b1  = hi ? 0.0f : b1y;
    float part = b0 * b0 + b1 * b1;
    float sq   = part + __shfl_xor(part, 16);   // |p|^2 for this lane's column
    v2f bfrag; bfrag.x = b0; bfrag.y = b1;
    v8f c = {0.f, 0.f, 0.f, 0.f, 0.f, 0.f, 0.f, 0.f};
    c = __builtin_amdgcn_wmma_f32_16x16x4_f32(false, afrag, false, bfrag,
                                              (short)0, c, false, false);
#pragma unroll
    for (int r = 0; r < 8; r++)
      keytile[r + 8 * hi][ln] = sq - 2.0f * c[r];
    __builtin_amdgcn_wave_barrier();     // scheduling fence only; DS is in-order
#pragma unroll
    for (int j = 0; j < 8; j++) {
      float kv = keytile[ln][8 * hi + j];
      int   mi = m0 + 8 * hi + j;
      if (kv < bestk[KNB - 1]) {
        bestk[KNB - 1] = kv; besti[KNB - 1] = mi;
#pragma unroll
        for (int i = KNB - 1; i > 0; i--) {
          if (bestk[i] < bestk[i - 1]) {
            float tf = bestk[i]; bestk[i] = bestk[i - 1]; bestk[i - 1] = tf;
            int   ti = besti[i]; besti[i] = besti[i - 1]; besti[i - 1] = ti;
          }
        }
      }
    }
    __builtin_amdgcn_wave_barrier();
  }

  // merge the two half-lists per row (each sorted ascending)
#pragma unroll
  for (int i = 0; i < KNB; i++) {
    mgk[ln][hi * KNB + i] = bestk[i];
    mgi[ln][hi * KNB + i] = besti[i];
  }
  __builtin_amdgcn_wave_barrier();
  if (hi == 0) {
    int ia = 0, ib = KNB;
    int* dst = knn_out + ((size_t)(b * N_) + q0 + ln) * KNB;
#pragma unroll
    for (int j = 0; j < KNB; j++) {
      float ka = (ia < KNB)     ? mgk[ln][ia] : 3.0e38f;
      float kb = (ib < 2 * KNB) ? mgk[ln][ib] : 3.0e38f;
      int sel;
      if (ka <= kb) { sel = mgi[ln][ia]; ia++; }
      else          { sel = mgi[ln][ib]; ib++; }
      dst[j] = sel;
    }
  }
}

// ---------------------------------------------------------------------------
// 3) factored GEMMs via V_WMMA_F32_16X16X32_F16 (K=8 padded to 32).
//    Amat[b][n][o] = sum_c  w1[o,c]        * fm[b,c,n]
//    Bmat[b][n][o] = sum_c (w1[o,c+8]-w1[o,c]) * fm[b,c,n]
//    Then h[b,o,n,k] = Amat[b][idx[n,k]][o] + Bmat[b][n][o]
// ---------------------------------------------------------------------------
__global__ void k_gemm(const float* __restrict__ fm, const float* __restrict__ w1,
                       float* __restrict__ Amat, float* __restrict__ Bmat) {
  const int b      = blockIdx.x;
  const int nchunk = blockIdx.y;
  const int t      = threadIdx.x;
  const int wave   = t >> 5;
  const int lane   = t & 31;
  const int ln     = lane & 15;
  const int hi     = lane >> 4;
  const int o      = wave * 16 + ln;   // output channel row

  // A fragments (weights, 16x32): nonzero only lanes 0-15, halves 0..7 (K=c<8)
  v16h fwa, fwb;
#pragma unroll
  for (int e = 0; e < 16; e++) { fwa[e] = (_Float16)0.0f; fwb[e] = (_Float16)0.0f; }
#pragma unroll
  for (int e = 0; e < 8; e++) {
    float wa = w1[o * 16 + e];
    float wb = w1[o * 16 + 8 + e] - wa;
    fwa[e] = hi ? (_Float16)0.0f : (_Float16)wa;
    fwb[e] = hi ? (_Float16)0.0f : (_Float16)wb;
  }

  for (int nt = 0; nt < 16; nt++) {
    const int n0 = nchunk * 256 + nt * 16;
    const int n  = n0 + ln;
    // B fragment (fm, 32x16): lanes 0-15 hold K=0..15 of column n; K>=8 zero
    v16h fb;
#pragma unroll
    for (int e = 0; e < 16; e++) fb[e] = (_Float16)0.0f;
#pragma unroll
    for (int e = 0; e < 8; e++) {
      float v = fm[((size_t)(b * CMIDH + e)) * N_ + n];
      fb[e] = hi ? (_Float16)0.0f : (_Float16)v;
    }
    v8f cA = {0.f, 0.f, 0.f, 0.f, 0.f, 0.f, 0.f, 0.f};
    v8f cB = {0.f, 0.f, 0.f, 0.f, 0.f, 0.f, 0.f, 0.f};
    cA = __builtin_amdgcn_wmma_f32_16x16x32_f16(false, fwa, false, fb, (short)0, cA, false, false);
    cB = __builtin_amdgcn_wmma_f32_16x16x32_f16(false, fwb, false, fb, (short)0, cB, false, false);
    const int obase = wave * 16 + 8 * hi;       // rows M = r + 8*hi
    float* pa = Amat + ((size_t)(b * N_) + n) * COUT + obase;
    float* pb = Bmat + ((size_t)(b * N_) + n) * COUT + obase;
    float4 a0 = {cA[0], cA[1], cA[2], cA[3]};
    float4 a1 = {cA[4], cA[5], cA[6], cA[7]};
    float4 b0 = {cB[0], cB[1], cB[2], cB[3]};
    float4 b1 = {cB[4], cB[5], cB[6], cB[7]};
    *(float4*)(pa)     = a0;
    *(float4*)(pa + 4) = a1;
    *(float4*)(pb)     = b0;
    *(float4*)(pb + 4) = b1;
  }
}

// ---------------------------------------------------------------------------
// 4) group-norm statistics: per (b,g) partial sums over 32 chans × N × K
//    float4 gathers: rows of Amat/Bmat are 32 float4s; lanes cover consecutive
//    chunks so each 8-lane group pulls contiguous 128B per gathered row.
// ---------------------------------------------------------------------------
__global__ void k_stats(const float* __restrict__ Amat, const float* __restrict__ Bmat,
                        const int* __restrict__ knn, float* __restrict__ part) {
  const int b = blockIdx.x, g = blockIdx.y, ch = blockIdx.z;
  const int t = threadIdx.x;
  const int o4 = t & 7;                       // float4 chunk within the group
  const float4* Am4 = (const float4*)Amat;
  const float4* Bm4 = (const float4*)Bmat;
  float s1 = 0.f, s2 = 0.f;
  for (int i = (t >> 3) & 31; i < 128; i += 32) {
    const int n = ch * 128 + i;
    const size_t rowB = ((size_t)(b * N_) + n) * 32 + g * 8 + o4;
    const float4 bv = Bm4[rowB];
    const int* kn = knn + ((size_t)(b * N_) + n) * KNB;
#pragma unroll
    for (int k = 0; k < KNB; k++) {
      int m = kn[k];
      float4 a = Am4[((size_t)(b * N_) + m) * 32 + g * 8 + o4];
      float h0 = a.x + bv.x, h1 = a.y + bv.y, h2 = a.z + bv.z, h3 = a.w + bv.w;
      s1 += (h0 + h1) + (h2 + h3);
      s2 += (h0 * h0 + h1 * h1) + (h2 * h2 + h3 * h3);
    }
  }
#pragma unroll
  for (int off = 16; off >= 1; off >>= 1) { s1 += __shfl_xor(s1, off); s2 += __shfl_xor(s2, off); }
  __shared__ float l1[8], l2[8];
  if ((t & 31) == 0) { l1[t >> 5] = s1; l2[t >> 5] = s2; }
  __syncthreads();
  if (t == 0) {
    float a = 0.f, c2 = 0.f;
    for (int w = 0; w < 8; w++) { a += l1[w]; c2 += l2[w]; }
    size_t idx = (((size_t)b * NG + g) * 32 + ch) * 2;
    part[idx] = a; part[idx + 1] = c2;
  }
}

__global__ void k_stats2(const float* __restrict__ part, float* __restrict__ stats) {
  int t = threadIdx.x;
  if (t >= B_ * NG) return;
  float s1 = 0.f, s2 = 0.f;
  for (int ch = 0; ch < 32; ch++) {
    s1 += part[((size_t)t * 32 + ch) * 2];
    s2 += part[((size_t)t * 32 + ch) * 2 + 1];
  }
  const float cnt = (float)(32 * N_ * KNB);   // (COUT/NG)*N*K
  float mu  = s1 / cnt;
  float var = s2 / cnt - mu * mu;
  stats[t * 2]     = mu;
  stats[t * 2 + 1] = rsqrtf(var + EPS_);
}

// ---------------------------------------------------------------------------
// 5) farthest point sampling: one 1024-thread block per batch, dist in regs
// ---------------------------------------------------------------------------
__global__ void __launch_bounds__(1024) k_fps(const float* __restrict__ x,
                                              int* __restrict__ fps, float* __restrict__ out) {
  const int b = blockIdx.x;
  const int t = threadIdx.x;
  const float* xb = x + (size_t)b * 3 * N_;
  float px[4], py[4], pz[4], dist[4];
#pragma unroll
  for (int j = 0; j < 4; j++) {
    int n = t + j * 1024;
    px[j] = xb[n]; py[j] = xb[N_ + n]; pz[j] = xb[2 * N_ + n];
  }
  float fx = xb[0], fy = xb[N_], fz = xb[2 * N_];
#pragma unroll
  for (int j = 0; j < 4; j++) {
    float dx = px[j] - fx, dy = py[j] - fy, dz = pz[j] - fz;
    dist[j] = dx * dx + dy * dy + dz * dz;
  }
  __shared__ int   sel_lds[NP];
  __shared__ float swv[32];
  __shared__ int   swi[32];
  __shared__ float selc[3];
  if (t == 0) sel_lds[0] = 0;

  for (int s = 1; s < NP; s++) {
    float bv = -1.0f; int bi = 0;
#pragma unroll
    for (int j = 0; j < 4; j++) {
      if (dist[j] > bv) { bv = dist[j]; bi = t + j * 1024; }
    }
#pragma unroll
    for (int off = 16; off >= 1; off >>= 1) {
      float ov = __shfl_xor(bv, off); int oi = __shfl_xor(bi, off);
      if (ov > bv || (ov == bv && oi < bi)) { bv = ov; bi = oi; }
    }
    if ((t & 31) == 0) { swv[t >> 5] = bv; swi[t >> 5] = bi; }
    __syncthreads();
    if (t < 32) {
      bv = swv[t]; bi = swi[t];
#pragma unroll
      for (int off = 16; off >= 1; off >>= 1) {
        float ov = __shfl_xor(bv, off); int oi = __shfl_xor(bi, off);
        if (ov > bv || (ov == bv && oi < bi)) { bv = ov; bi = oi; }
      }
      if (t == 0) {
        sel_lds[s] = bi;
        selc[0] = xb[bi]; selc[1] = xb[N_ + bi]; selc[2] = xb[2 * N_ + bi];
      }
    }
    __syncthreads();
    float sx = selc[0], sy = selc[1], sz = selc[2];
#pragma unroll
    for (int j = 0; j < 4; j++) {
      float dx = px[j] - sx, dy = py[j] - sy, dz = pz[j] - sz;
      float d = dx * dx + dy * dy + dz * dz;
      dist[j] = fminf(dist[j], d);
    }
  }
  __syncthreads();
  // emit indices (ws + d_out as float) and gathered coordinates
  {
    int s  = t;
    int id = sel_lds[s];
    fps[(size_t)b * NP + s] = id;
    out[OUT_IDX + (size_t)b * NP + s] = (float)id;
    out[(size_t)b * 3 * NP + 0 * NP + s] = xb[id];
    out[(size_t)b * 3 * NP + 1 * NP + s] = xb[N_ + id];
    out[(size_t)b * 3 * NP + 2 * NP + s] = xb[2 * N_ + id];
  }
}

// ---------------------------------------------------------------------------
// 6) finalize: normalize + leaky-relu + max over k, only at sampled points.
//    Each thread handles 4 consecutive channels (float4 gathers, coalesced).
// ---------------------------------------------------------------------------
__global__ void k_final(const float* __restrict__ Amat, const float* __restrict__ Bmat,
                        const int* __restrict__ knn, const int* __restrict__ fps,
                        const float* __restrict__ stats, const float* __restrict__ gn_w,
                        const float* __restrict__ gn_b, float* __restrict__ out) {
  int gid = blockIdx.x * blockDim.x + threadIdx.x;   // B*NP*32 threads
  int oq = gid & 31;                                 // float4 chunk (o = oq*4)
  int s  = (gid >> 5) & (NP - 1);
  int b  = gid >> 15;
  int n  = fps[(size_t)b * NP + s];
  const float4* Am4 = (const float4*)Amat;
  const float4* Bm4 = (const float4*)Bmat;
  float4 bv = Bm4[((size_t)(b * N_) + n) * 32 + oq];
  int g = oq >> 3;
  float mu = stats[(b * NG + g) * 2];
  float rs = stats[(b * NG + g) * 2 + 1];
  float4 w4 = ((const float4*)gn_w)[oq];
  float4 b4 = ((const float4*)gn_b)[oq];
  const int* kn = knn + ((size_t)(b * N_) + n) * KNB;
  float best0 = -3.4e38f, best1 = -3.4e38f, best2 = -3.4e38f, best3 = -3.4e38f;
#pragma unroll
  for (int k = 0; k < KNB; k++) {
    int m = kn[k];
    float4 a = Am4[((size_t)(b * N_) + m) * 32 + oq];
    float h0 = a.x + bv.x, h1 = a.y + bv.y, h2 = a.z + bv.z, h3 = a.w + bv.w;
    float y0 = (h0 - mu) * rs * w4.x + b4.x;  y0 = (y0 >= 0.f) ? y0 : SLOPE * y0;
    float y1 = (h1 - mu) * rs * w4.y + b4.y;  y1 = (y1 >= 0.f) ? y1 : SLOPE * y1;
    float y2 = (h2 - mu) * rs * w4.z + b4.z;  y2 = (y2 >= 0.f) ? y2 : SLOPE * y2;
    float y3 = (h3 - mu) * rs * w4.w + b4.w;  y3 = (y3 >= 0.f) ? y3 : SLOPE * y3;
    best0 = fmaxf(best0, y0); best1 = fmaxf(best1, y1);
    best2 = fmaxf(best2, y2); best3 = fmaxf(best3, y3);
  }
  size_t obase = (size_t)b * COUT + oq * 4;
  out[OUT_FEAT + (obase + 0) * NP + s] = best0;
  out[OUT_FEAT + (obase + 1) * NP + s] = best1;
  out[OUT_FEAT + (obase + 2) * NP + s] = best2;
  out[OUT_FEAT + (obase + 3) * NP + s] = best3;
}

// ---------------------------------------------------------------------------
extern "C" void kernel_launch(void* const* d_in, const int* in_sizes, int n_in,
                              void* d_out, int out_size, void* d_ws, size_t ws_size,
                              hipStream_t stream) {
  const float* x    = (const float*)d_in[0];
  const float* f    = (const float*)d_in[1];
  const float* w_in = (const float*)d_in[2];
  const float* b_in = (const float*)d_in[3];
  const float* w1   = (const float*)d_in[4];
  const float* gn_w = (const float*)d_in[5];
  const float* gn_b = (const float*)d_in[6];
  float* out = (float*)d_out;

  // workspace carve-up (~37 MB)
  float* fm    = (float*)d_ws;
  float* Amat  = fm + (size_t)B_ * CMIDH * N_;
  float* Bmat  = Amat + (size_t)B_ * N_ * COUT;
  int*   knn   = (int*)(Bmat + (size_t)B_ * N_ * COUT);
  int*   fps   = knn + (size_t)B_ * N_ * KNB;
  float* part  = (float*)(fps + (size_t)B_ * NP);
  float* stats = part + (size_t)B_ * NG * 32 * 2;

  k_lift  <<<dim3((B_ * N_) / 256), 256, 0, stream>>>(f, w_in, b_in, fm);
  k_knn   <<<dim3(N_ / 16, B_),      32, 0, stream>>>(x, knn);
  k_gemm  <<<dim3(B_, N_ / 256),    256, 0, stream>>>(fm, w1, Amat, Bmat);
  k_stats <<<dim3(B_, NG, 32),      256, 0, stream>>>(Amat, Bmat, knn, part);
  k_stats2<<<1, 32, 0, stream>>>(part, stats);
  k_fps   <<<B_, 1024, 0, stream>>>(x, fps, out);
  k_final <<<(B_ * NP * 32) / 256, 256, 0, stream>>>(Amat, Bmat, knn, fps, stats, gn_w, gn_b, out);

  (void)in_sizes; (void)n_in; (void)out_size; (void)ws_size;
}
